// LOCOSpectralConv2D_18365280158085
// MI455X (gfx1250) — matrix operand order repaired and checked by reading, not compiled, not run on hardware
//
#include <hip/hip_runtime.h>
#include <math.h>

typedef _Float16 f16;
typedef __attribute__((ext_vector_type(16))) _Float16 v16h;
typedef __attribute__((ext_vector_type(8)))  float    v8f;

#define WMMA(a, b, c) \
  __builtin_amdgcn_wmma_f32_16x16x32_f16(false, (a), false, (b), (short)0, (c), false, false)

// Problem constants
#define NPIX   256     // H == W
#define NM     16      // retained modes per dim
#define NCH    64      // CIN == COUT
#define NB     8       // batch
#define NPAIR  512     // NB * NCH
#define TWOPI  6.28318530717958647692f

// A-matrix (16x32 f16) K index for vgpr-pair v (0..7), lane half h
__device__ __forceinline__ int a_k(int v, int h) {
  return (v < 4) ? (8 * h + 2 * v) : (16 + 8 * h + 2 * (v - 4));
}
// B-matrix (32x16 f16) K index
__device__ __forceinline__ int b_k(int v, int h) { return (v < 4) ? (8 * h + 2 * v) : (16 + 8 * h + 2 * (v - 4)); }

__device__ __forceinline__ float gelu_exact(float x) {
  return 0.5f * x * (1.0f + erff(x * 0.70710678118654752f));
}

// ---------------------------------------------------------------------------
// K0: twiddle tables. Cr[k][h] = cos(2*pi*k*h/256)/16 ; Sn = sin(...)/16
// (the 1/16 is the per-1D-stage ortho norm, applied once per DFT stage)
// ---------------------------------------------------------------------------
__global__ void k0_tables(float* __restrict__ Cr, float* __restrict__ Sn) {
  int idx = blockIdx.x * 256 + threadIdx.x;   // 16 blocks -> 4096 entries
  int k = idx >> 8, h = idx & 255;
  int ph = (k * h) & 255;                     // exact periodic reduction
  float ang = TWOPI * (float)ph * (1.0f / 256.0f);
  float s, c;
  __sincosf(ang, &s, &c);
  Cr[idx] = c * 0.0625f;
  Sn[idx] = s * 0.0625f;
}

// ---------------------------------------------------------------------------
// K1: partial forward DFT over H.  tmp[bc][kx][w] = sum_h F[kx,h]*u[bc][h][w]
// F = Cr - i*Sn.  One wave per (bc, 16-wide w tile). 8192 wave tasks.
// ---------------------------------------------------------------------------
__global__ void k1_fwd_h(const float* __restrict__ u,
                         const float* __restrict__ Cr, const float* __restrict__ Sn,
                         float* __restrict__ tr, float* __restrict__ ti) {
  const int lane = threadIdx.x & 31, wave = threadIdx.x >> 5;
  const int lm = lane & 15, hf = lane >> 4;
  const int task = blockIdx.x * 8 + wave;        // 0..8191
  const int bc = task >> 4, w0 = (task & 15) * 16;
  const float* ub = u + (size_t)bc * (NPIX * NPIX);

  v8f accR = {0.f, 0.f, 0.f, 0.f, 0.f, 0.f, 0.f, 0.f};
  v8f accI = accR;
  for (int kc = 0; kc < 8; ++kc) {
    const int h0 = kc * 32;
    v16h aC, aS, bU;
#pragma unroll
    for (int v = 0; v < 8; ++v) {
      const int ka = a_k(v, hf);
      aC[2 * v]     = (f16)Cr[lm * NPIX + h0 + ka];
      aC[2 * v + 1] = (f16)Cr[lm * NPIX + h0 + ka + 1];
      aS[2 * v]     = (f16)(-Sn[lm * NPIX + h0 + ka]);      // F_i = -sin
      aS[2 * v + 1] = (f16)(-Sn[lm * NPIX + h0 + ka + 1]);
      const int kb = b_k(v, hf);
      bU[2 * v]     = (f16)ub[(h0 + kb) * NPIX + w0 + lm];
      bU[2 * v + 1] = (f16)ub[(h0 + kb + 1) * NPIX + w0 + lm];
    }
    accR = WMMA(aC, bU, accR);
    accI = WMMA(aS, bU, accI);
  }
#pragma unroll
  for (int v = 0; v < 8; ++v) {
    const int kx = v + 8 * hf;
    tr[(size_t)bc * (NM * NPIX) + kx * NPIX + w0 + lm] = accR[v];
    ti[(size_t)bc * (NM * NPIX) + kx * NPIX + w0 + lm] = accI[v];
  }
}

// ---------------------------------------------------------------------------
// K2: forward DFT over W. u_low[bc][kx][ky] = sum_w tmp[kx,w]*(Cr[ky,w]-i*Sn[ky,w])
// One wave per bc. 512 wave tasks.
// ---------------------------------------------------------------------------
__global__ void k2_fwd_w(const float* __restrict__ tr, const float* __restrict__ ti,
                         const float* __restrict__ Cr, const float* __restrict__ Sn,
                         float* __restrict__ ulr, float* __restrict__ uli) {
  const int lane = threadIdx.x & 31, wave = threadIdx.x >> 5;
  const int lm = lane & 15, hf = lane >> 4;
  const int bc = blockIdx.x * 8 + wave;          // 0..511
  const float* trb = tr + (size_t)bc * (NM * NPIX);
  const float* tib = ti + (size_t)bc * (NM * NPIX);

  v8f accR = {0.f, 0.f, 0.f, 0.f, 0.f, 0.f, 0.f, 0.f};
  v8f accI = accR;
  for (int kc = 0; kc < 8; ++kc) {
    const int w0 = kc * 32;
    v16h aR, aI, aRn, bC, bS;
#pragma unroll
    for (int v = 0; v < 8; ++v) {
      const int ka = a_k(v, hf);
      const float r0 = trb[lm * NPIX + w0 + ka], r1 = trb[lm * NPIX + w0 + ka + 1];
      const float i0 = tib[lm * NPIX + w0 + ka], i1 = tib[lm * NPIX + w0 + ka + 1];
      aR[2 * v] = (f16)r0;  aR[2 * v + 1] = (f16)r1;
      aI[2 * v] = (f16)i0;  aI[2 * v + 1] = (f16)i1;
      aRn[2 * v] = (f16)(-r0); aRn[2 * v + 1] = (f16)(-r1);
      const int kb = b_k(v, hf);
      bC[2 * v]     = (f16)Cr[lm * NPIX + w0 + kb];       // B[k][n] = Cr[n][w0+k]
      bC[2 * v + 1] = (f16)Cr[lm * NPIX + w0 + kb + 1];
      bS[2 * v]     = (f16)Sn[lm * NPIX + w0 + kb];
      bS[2 * v + 1] = (f16)Sn[lm * NPIX + w0 + kb + 1];
    }
    accR = WMMA(aR, bC, accR);   // +tr*C
    accR = WMMA(aI, bS, accR);   // +ti*S
    accI = WMMA(aI, bC, accI);   // +ti*C
    accI = WMMA(aRn, bS, accI);  // -tr*S
  }
#pragma unroll
  for (int v = 0; v < 8; ++v) {
    const int kx = v + 8 * hf;
    ulr[bc * (NM * NM) + kx * NM + lm] = accR[v];
    uli[bc * (NM * NM) + kx * NM + lm] = accI[v];
  }
}

// ---------------------------------------------------------------------------
// K3: per-mode complex channel mix + exact GELU.
// A(16x64) rows 0..7 = u_low_re[b], rows 8..15 = u_low_im[b].  P = A@w_re, Q = A@w_im.
// Wu_re[b] = P(half0) - Q(half1);  Wu_im[b] = Q(half0) + P(half1)  -> ds_swizzle SWAPX16.
// One wave per (mode, 16-wide o tile). 1024 wave tasks.
// ---------------------------------------------------------------------------
__global__ void k3_mix(const float* __restrict__ ulr, const float* __restrict__ uli,
                       const float* __restrict__ wre, const float* __restrict__ wim,
                       float* __restrict__ sr, float* __restrict__ si) {
  const int lane = threadIdx.x & 31, wave = threadIdx.x >> 5;
  const int lm = lane & 15, hf = lane >> 4;
  const int task = blockIdx.x * 8 + wave;       // 0..1023
  const int mode = task >> 2, ot = task & 3;
  const int o = ot * 16 + lm;
  const int b = lm & 7, pl = lm >> 3;
  const float* usrc = pl ? uli : ulr;

  v8f P = {0.f, 0.f, 0.f, 0.f, 0.f, 0.f, 0.f, 0.f};
  v8f Q = P;
  for (int kc = 0; kc < 2; ++kc) {
    v16h aU, bR, bI;
#pragma unroll
    for (int v = 0; v < 8; ++v) {
      const int ka = kc * 32 + a_k(v, hf);
      aU[2 * v]     = (f16)usrc[(b * NCH + ka) * (NM * NM) + mode];
      aU[2 * v + 1] = (f16)usrc[(b * NCH + ka + 1) * (NM * NM) + mode];
      const int kb = kc * 32 + b_k(v, hf);
      bR[2 * v]     = (f16)wre[((size_t)mode * NCH + kb) * NCH + o];
      bR[2 * v + 1] = (f16)wre[((size_t)mode * NCH + kb + 1) * NCH + o];
      bI[2 * v]     = (f16)wim[((size_t)mode * NCH + kb) * NCH + o];
      bI[2 * v + 1] = (f16)wim[((size_t)mode * NCH + kb + 1) * NCH + o];
    }
    P = WMMA(aU, bR, P);
    Q = WMMA(aU, bI, Q);
  }
  float* dst = hf ? si : sr;
#pragma unroll
  for (int v = 0; v < 8; ++v) {
    // swap lane halves: lane l <-> l^16 (group-of-32 swizzle, xor=0x10,and=0x1f)
    const float qs = __int_as_float(
        __builtin_amdgcn_ds_swizzle(__float_as_int(Q[v]), 0x401F));
    const float val = hf ? (P[v] + qs) : (P[v] - qs);
    dst[(v * NCH + o) * (NM * NM) + mode] = gelu_exact(val);
  }
}

// ---------------------------------------------------------------------------
// K4: 16x16 complex circular convolution: conv = circconv(u_low, sigma) / 256.
// One 256-thread block per (b,o) pair. 512 blocks.
// The four 1KB input arrays are staged into LDS with CDNA5 async loads
// (GLOBAL_LOAD_ASYNC_TO_LDS_B128, ASYNCcnt-tracked): one 16B transfer/thread.
// ---------------------------------------------------------------------------
__global__ void k4_circconv(const float* __restrict__ ulr, const float* __restrict__ uli,
                            const float* __restrict__ sr, const float* __restrict__ si,
                            float* __restrict__ cvr, float* __restrict__ cvi) {
  __shared__ float buf[1024];                 // [0:256)=Ar [256:512)=Ai [512:768)=Br [768:1024)=Bi
  const int bo = blockIdx.x, t = threadIdx.x;

  // thread t stages buf[4t .. 4t+3]: array a = t>>6, element (t&63)*4
  const float* s01 = (t & 64) ? uli : ulr;
  const float* s23 = (t & 64) ? si : sr;
  const float* src = (t & 128) ? s23 : s01;
  const unsigned long long gaddr =
      (unsigned long long)(uintptr_t)(src + bo * 256 + (t & 63) * 4);
  // low 32 bits of a generic LDS pointer are the LDS byte offset (ISA 10.2)
  const unsigned ldsoff = (unsigned)(uintptr_t)(&buf[0]) + (unsigned)t * 16u;
  asm volatile("global_load_async_to_lds_b128 %0, %1, off"
               :: "v"(ldsoff), "v"(gaddr) : "memory");
  asm volatile("s_wait_asynccnt 0x0" ::: "memory");
  __syncthreads();

  const int x = t >> 4, y = t & 15;
  float cr = 0.f, ci = 0.f;
  for (int p = 0; p < 16; ++p) {
    const int xp = ((x - p) & 15) * 16;
#pragma unroll
    for (int q = 0; q < 16; ++q) {
      const float ar = buf[p * 16 + q], ai = buf[256 + p * 16 + q];
      const int sidx = xp + ((y - q) & 15);
      const float br = buf[512 + sidx], bi = buf[768 + sidx];
      cr = fmaf(ar, br, cr); cr = fmaf(-ai, bi, cr);
      ci = fmaf(ar, bi, ci); ci = fmaf(ai, br, ci);
    }
  }
  cvr[bo * 256 + t] = cr * (1.0f / 256.0f);
  cvi[bo * 256 + t] = ci * (1.0f / 256.0f);
}

// ---------------------------------------------------------------------------
// K5a: inverse DFT over rows:  t[h,y] = sum_x (Cr[x,h]+i*Sn[x,h]) * conv[x,y]
// t_re = [CrT | -SnT](256x32) @ [cv_r; cv_i] ; t_im = [CrT | SnT] @ [cv_i; cv_r]
// One wave per (bo, 16-row h tile). 8192 wave tasks.
// ---------------------------------------------------------------------------
__global__ void k5a_inv_h(const float* __restrict__ cvr, const float* __restrict__ cvi,
                          const float* __restrict__ Cr, const float* __restrict__ Sn,
                          float* __restrict__ tr, float* __restrict__ ti) {
  const int lane = threadIdx.x & 31, wave = threadIdx.x >> 5;
  const int lm = lane & 15, hf = lane >> 4;
  const int task = blockIdx.x * 8 + wave;       // 0..8191
  const int bo = task >> 4, h0 = (task & 15) * 16;

  v16h aR, aI, bRI, bIR;
#pragma unroll
  for (int v = 0; v < 8; ++v) {
    const int ka = a_k(v, hf);
#pragma unroll
    for (int e = 0; e < 2; ++e) {
      const int k = ka + e;
      float re, ri;
      if (k < 16) {
        const float c = Cr[k * NPIX + h0 + lm];
        re = c; ri = c;
      } else {
        const float s = Sn[(k - 16) * NPIX + h0 + lm];
        re = -s; ri = s;
      }
      aR[2 * v + e] = (f16)re;
      aI[2 * v + e] = (f16)ri;
    }
    const int kb = b_k(v, hf);
#pragma unroll
    for (int e = 0; e < 2; ++e) {
      const int k = kb + e;
      float vri, vir;
      if (k < 16) {
        vri = cvr[bo * 256 + k * 16 + lm];
        vir = cvi[bo * 256 + k * 16 + lm];
      } else {
        vri = cvi[bo * 256 + (k - 16) * 16 + lm];
        vir = cvr[bo * 256 + (k - 16) * 16 + lm];
      }
      bRI[2 * v + e] = (f16)vri;
      bIR[2 * v + e] = (f16)vir;
    }
  }
  v8f z = {0.f, 0.f, 0.f, 0.f, 0.f, 0.f, 0.f, 0.f};
  v8f accR = WMMA(aR, bRI, z);
  v8f accI = WMMA(aI, bIR, z);
#pragma unroll
  for (int v = 0; v < 8; ++v) {
    const int h = h0 + v + 8 * hf;
    tr[(size_t)bo * (NPIX * NM) + h * NM + lm] = accR[v];
    ti[(size_t)bo * (NPIX * NM) + h * NM + lm] = accI[v];
  }
}

// ---------------------------------------------------------------------------
// K5b: inverse DFT over cols, real output:
// out[h,w] = sum_y t_r[h,y]*Cr[y,w] - t_i[h,y]*Sn[y,w]
//          = [t_r | -t_i](256x32) @ [Cr; Sn](32x256)
// One wave per (bo, h tile); each wave sweeps 16 w tiles reusing its A frag.
// 8192 wave tasks. Writes the final 134 MB output (store-bound).
// ---------------------------------------------------------------------------
__global__ void k5b_inv_w(const float* __restrict__ tr, const float* __restrict__ ti,
                          const float* __restrict__ Cr, const float* __restrict__ Sn,
                          float* __restrict__ out) {
  const int lane = threadIdx.x & 31, wave = threadIdx.x >> 5;
  const int lm = lane & 15, hf = lane >> 4;
  const int task = blockIdx.x * 8 + wave;       // 0..8191
  const int bo = task >> 4, h0 = (task & 15) * 16;
  const float* trb = tr + (size_t)bo * (NPIX * NM);
  const float* tib = ti + (size_t)bo * (NPIX * NM);

  // hint the A-operand rows into cache (global_prefetch_b8)
  __builtin_prefetch(&trb[(h0 + lm) * NM], 0, 1);
  __builtin_prefetch(&tib[(h0 + lm) * NM], 0, 1);

  v16h aT;
#pragma unroll
  for (int v = 0; v < 8; ++v) {
    const int ka = a_k(v, hf);
#pragma unroll
    for (int e = 0; e < 2; ++e) {
      const int k = ka + e;
      const float val = (k < 16) ? trb[(h0 + lm) * NM + k]
                                 : -tib[(h0 + lm) * NM + (k - 16)];
      aT[2 * v + e] = (f16)val;
    }
  }
  for (int nt = 0; nt < 16; ++nt) {
    const int w0 = nt * 16;
    v16h bE;
#pragma unroll
    for (int v = 0; v < 8; ++v) {
      const int kb = b_k(v, hf);
#pragma unroll
      for (int e = 0; e < 2; ++e) {
        const int k = kb + e;
        const float val = (k < 16) ? Cr[k * NPIX + w0 + lm]
                                   : Sn[(k - 16) * NPIX + w0 + lm];
        bE[2 * v + e] = (f16)val;
      }
    }
    v8f acc = {0.f, 0.f, 0.f, 0.f, 0.f, 0.f, 0.f, 0.f};
    acc = WMMA(aT, bE, acc);
#pragma unroll
    for (int v = 0; v < 8; ++v) {
      const int h = h0 + v + 8 * hf;
      out[(size_t)bo * (NPIX * NPIX) + (size_t)h * NPIX + w0 + lm] = acc[v];
    }
  }
}

// ---------------------------------------------------------------------------
extern "C" void kernel_launch(void* const* d_in, const int* in_sizes, int n_in,
                              void* d_out, int out_size, void* d_ws, size_t ws_size,
                              hipStream_t stream) {
  const float* u   = (const float*)d_in[0];
  const float* wre = (const float*)d_in[1];
  const float* wim = (const float*)d_in[2];
  float* out = (float*)d_out;

  // Workspace layout (floats). Total = 9,183,232 floats (~36.7 MB).
  float* ws   = (float*)d_ws;
  float* Cr   = ws;                 // 16*256
  float* Sn   = Cr + 4096;          // 16*256
  float* tmpr = Sn + 4096;          // 512*16*256
  float* tmpi = tmpr + 2097152;
  float* ulr  = tmpi + 2097152;     // 512*256
  float* uli  = ulr + 131072;
  float* sr   = uli + 131072;
  float* si   = sr + 131072;
  float* cvr  = si + 131072;
  float* cvi  = cvr + 131072;
  float* t2r  = cvi + 131072;       // 512*256*16
  float* t2i  = t2r + 2097152;

  dim3 blk(256);
  k0_tables <<<16,   blk, 0, stream>>>(Cr, Sn);
  k1_fwd_h  <<<1024, blk, 0, stream>>>(u, Cr, Sn, tmpr, tmpi);
  k2_fwd_w  <<<64,   blk, 0, stream>>>(tmpr, tmpi, Cr, Sn, ulr, uli);
  k3_mix    <<<128,  blk, 0, stream>>>(ulr, uli, wre, wim, sr, si);
  k4_circconv<<<512, blk, 0, stream>>>(ulr, uli, sr, si, cvr, cvi);
  k5a_inv_h <<<1024, blk, 0, stream>>>(cvr, cvi, Cr, Sn, t2r, t2i);
  k5b_inv_w <<<1024, blk, 0, stream>>>(t2r, t2i, Cr, Sn, out);
}
